// CausalPriorityAttention_89129161326847
// MI455X (gfx1250) — matrix operand order repaired and checked by reading, not compiled, not run on hardware
//
#include <hip/hip_runtime.h>

typedef __bf16 bf16;
typedef __attribute__((ext_vector_type(16))) __bf16 v16bf;
typedef __attribute__((ext_vector_type(8)))  __bf16 v8bf;
typedef __attribute__((ext_vector_type(8)))  float  v8f;

// D = A(16x32 bf16) * B(32x16 bf16) + C(16x16 f32), wave32
__device__ __forceinline__ v8f wmma_bf16(v16bf a, v16bf b, v8f c) {
  return __builtin_amdgcn_wmma_f32_16x16x32_bf16(false, a, false, b, (short)0, c,
                                                 false, false);
}

// A-fragment loader (16-bit A 16x32), contiguous k order used by this code (B frags match):
// lane m = lane&15, half = lane>>4; elems 0..15 -> K = half*16 + i.
// p = row pointer at k0.
__device__ __forceinline__ v16bf load_a_frag(const bf16* p, int half) {
  v8bf lo = *(const v8bf*)(p + half * 16);
  v8bf hi = *(const v8bf*)(p + half * 16 + 8);
  v16bf a;
#pragma unroll
  for (int i = 0; i < 8; ++i) { a[i] = lo[i]; a[i + 8] = hi[i]; }
  return a;
}

// ---------------------------------------------------------------- convert
__global__ void cvt_f32_bf16(const float* __restrict__ in, bf16* __restrict__ out,
                             int n) {
  int i = blockIdx.x * blockDim.x + threadIdx.x;
  if (i < n) out[i] = (bf16)in[i];
}

// ---------------------------------------------------------------- QKV GEMM
// qkv[r,c] = sum_k x[r,k]*Win[c,k] + bin[c]
// scatter: q,k -> (B,H,N,64) bf16 (row-major over hd, feeds A / K^T-B frags)
//          v   -> (B,H,64,N) bf16 (transposed: contiguous along keys, feeds PV B frags)
__global__ void __launch_bounds__(256)
qkv_gemm(const bf16* __restrict__ A, const bf16* __restrict__ W,
         const float* __restrict__ bias, bf16* __restrict__ qb,
         bf16* __restrict__ kb, bf16* __restrict__ vt) {
  const int lane = threadIdx.x & 31;
  const int wv   = threadIdx.x >> 5;
  const int wg   = blockIdx.x * 8 + wv;   // 12288 waves total
  const int mt   = wg / 24;               // 512 row tiles (M=8192)
  const int nt   = wg % 24;               // 24 col tiles of 64 (N=1536)
  const int half = lane >> 4;
  const int nl   = lane & 15;
  const bf16* arow = A + (size_t)(mt * 16 + nl) * 512;

  v8f acc[4];
#pragma unroll
  for (int t = 0; t < 4; ++t)
#pragma unroll
    for (int j = 0; j < 8; ++j) acc[t][j] = 0.f;

  for (int k0 = 0; k0 < 512; k0 += 32) {
    v16bf a = load_a_frag(arow + k0, half);
#pragma unroll
    for (int t = 0; t < 4; ++t) {
      const int col = nt * 64 + t * 16 + nl;           // B: lane = N column
      v16bf bw = *(const v16bf*)(W + (size_t)col * 512 + k0 + (lane & 16));
      acc[t] = wmma_bf16(a, bw, acc[t]);
    }
  }
#pragma unroll
  for (int t = 0; t < 4; ++t) {
    const int col = nt * 64 + t * 16 + nl;
    const float bv = bias[col];
    const int buf = col >> 9;        // 0:q 1:k 2:v  (uniform within a 16-col subtile)
    const int c   = col & 511;
    const int hh  = c >> 6;
    const int d   = c & 63;
#pragma unroll
    for (int j = 0; j < 8; ++j) {
      const int r  = mt * 16 + j + 8 * half;  // C layout: row = j + 8*half
      const int bb = r >> 10;
      const int nn = r & 1023;
      const bf16 val = (bf16)(acc[t][j] + bv);
      if (buf == 2) {
        // V transposed: (B,H,64,N)
        vt[(((size_t)(bb * 8 + hh)) * 64 + d) * 1024 + nn] = val;
      } else {
        bf16* dst = (buf == 0) ? qb : kb;
        dst[(((size_t)(bb * 8 + hh)) * 1024 + nn) * 64 + d] = val;
      }
    }
  }
}

// ---------------------------------------------------------------- attention
// One wave per (b, h, 16-query tile); flash-style online softmax over 32-key blocks.
__global__ void __launch_bounds__(256)
attn_kernel(const bf16* __restrict__ qb, const bf16* __restrict__ kb,
            const bf16* __restrict__ vt, const float* __restrict__ gbias,
            const float* __restrict__ bstr, bf16* __restrict__ ob) {
  __shared__ bf16 lds[8][16][32];
  const int lane = threadIdx.x & 31;
  const int wv   = threadIdx.x >> 5;
  const int idx  = blockIdx.x * 8 + wv;   // 4096 waves: B*H*(N/16)
  const int b    = idx >> 9;
  const int h    = (idx >> 6) & 7;
  const int qt   = idx & 63;
  const int half = lane >> 4;
  const int nl   = lane & 15;
  const int q0   = qt * 16;

  const float alpha = 1.f / (1.f + __expf(-bstr[0]));
  const float beta  = (1.f - alpha) * 0.5f - 0.5f;  // bias = 10*(alpha*sig + beta)
  const float inv_sqrt = 0.125f;                     // 1/sqrt(64)

  const bf16* qrow = qb + (((size_t)(b * 8 + h)) * 1024 + q0 + nl) * 64;
  const v16bf aq0 = load_a_frag(qrow, half);       // hd 0..31
  const v16bf aq1 = load_a_frag(qrow + 32, half);  // hd 32..63

  v8f O[4];
  float m_run[8], l_run[8];
#pragma unroll
  for (int t = 0; t < 4; ++t)
#pragma unroll
    for (int j = 0; j < 8; ++j) O[t][j] = 0.f;
#pragma unroll
  for (int j = 0; j < 8; ++j) { m_run[j] = -1e30f; l_run[j] = 0.f; }

  const bf16* kbase = kb + ((size_t)(b * 8 + h)) * 1024 * 64;   // (keys, hd)
  const bf16* vbase = vt + ((size_t)(b * 8 + h)) * 64 * 1024;   // (hd, keys)
  const float* gbp  = gbias + (size_t)b * 1024 * 1024 + q0 + half * 8;

  for (int blk = 0; blk < 32; ++blk) {
    const int key0 = blk * 32;
    if (blk + 1 < 32) {
      __builtin_prefetch(kbase + (size_t)(key0 + 32) * 64, 0, 0);
      __builtin_prefetch(vbase + (size_t)nl * 1024 + key0 + 32, 0, 0);
    }
    float S[2][8];
#pragma unroll
    for (int t = 0; t < 2; ++t) {
      const int key = key0 + t * 16 + nl;               // B: lane = key column
      const bf16* krow = kbase + (size_t)key * 64 + (lane & 16);
      v16bf bk0 = *(const v16bf*)(krow);                // hd 0..31  (K^T)
      v16bf bk1 = *(const v16bf*)(krow + 32);           // hd 32..63
      v8f z;
#pragma unroll
      for (int j = 0; j < 8; ++j) z[j] = 0.f;
      v8f sc = wmma_bf16(aq0, bk0, z);
      sc = wmma_bf16(aq1, bk1, sc);
      // graph bias: gb[b, key, q] contiguous in q (8 rows of this half)
      v8f g = *(const v8f*)(gbp + (size_t)key * 1024);
#pragma unroll
      for (int j = 0; j < 8; ++j) {
        const float sg = 1.f / (1.f + __expf(-g[j]));
        S[t][j] = sc[j] * inv_sqrt + 10.f * (alpha * sg + beta);
      }
    }
    // online softmax: rows live across 16-lane halves in C layout
    float corr[8];
#pragma unroll
    for (int j = 0; j < 8; ++j) {
      float mx = fmaxf(S[0][j], S[1][j]);
#pragma unroll
      for (int msk = 1; msk < 16; msk <<= 1) mx = fmaxf(mx, __shfl_xor(mx, msk, 32));
      const float mn = fmaxf(m_run[j], mx);
      corr[j] = __expf(m_run[j] - mn);
      m_run[j] = mn;
      const float p0 = __expf(S[0][j] - mn);
      const float p1 = __expf(S[1][j] - mn);
      S[0][j] = p0;
      S[1][j] = p1;
      float rs = p0 + p1;
#pragma unroll
      for (int msk = 1; msk < 16; msk <<= 1) rs += __shfl_xor(rs, msk, 32);
      l_run[j] = l_run[j] * corr[j] + rs;
    }
#pragma unroll
    for (int t = 0; t < 4; ++t)
#pragma unroll
      for (int j = 0; j < 8; ++j) O[t][j] *= corr[j];

    // probs: C layout -> LDS -> A layout (bf16)
#pragma unroll
    for (int t = 0; t < 2; ++t)
#pragma unroll
      for (int j = 0; j < 8; ++j)
        lds[wv][j + 8 * half][t * 16 + nl] = (bf16)S[t][j];
    asm volatile("s_wait_dscnt 0" ::: "memory");  // intra-wave LDS ordering
    v16bf pa;
#pragma unroll
    for (int i = 0; i < 8; ++i) pa[i] = lds[wv][nl][half * 16 + i];
#pragma unroll
    for (int i = 0; i < 8; ++i) pa[i + 8] = lds[wv][nl][half * 16 + 8 + i];

    // PV: K-dim = 32 keys, 4 hd subtiles; V transposed -> contiguous B frags
#pragma unroll
    for (int t = 0; t < 4; ++t) {
      const bf16* vrow = vbase + (size_t)(t * 16 + nl) * 1024 + key0 + (lane & 16);
      v16bf bv = *(const v16bf*)(vrow);
      O[t] = wmma_bf16(pa, bv, O[t]);
    }
  }
  // finalize: divide by row sums, store as (B, N, D) bf16 with D = h*64 + d
#pragma unroll
  for (int j = 0; j < 8; ++j) l_run[j] = 1.f / l_run[j];
#pragma unroll
  for (int t = 0; t < 4; ++t)
#pragma unroll
    for (int j = 0; j < 8; ++j) {
      const int q = q0 + j + 8 * half;
      ob[((size_t)b * 1024 + q) * 512 + h * 64 + t * 16 + nl] =
          (bf16)(O[t][j] * l_run[j]);
    }
}

// ---------------------------------------------------------------- out-proj GEMM
__global__ void __launch_bounds__(256)
out_gemm(const bf16* __restrict__ A, const bf16* __restrict__ W,
         const float* __restrict__ bias, float* __restrict__ out) {
  const int lane = threadIdx.x & 31;
  const int wv   = threadIdx.x >> 5;
  const int wg   = blockIdx.x * 8 + wv;  // 4096 waves
  const int mt   = wg >> 3;              // 512 row tiles
  const int nt   = wg & 7;               // 8 col tiles of 64 (N=512)
  const int half = lane >> 4;
  const int nl   = lane & 15;
  const bf16* arow = A + (size_t)(mt * 16 + nl) * 512;

  v8f acc[4];
#pragma unroll
  for (int t = 0; t < 4; ++t)
#pragma unroll
    for (int j = 0; j < 8; ++j) acc[t][j] = 0.f;

  for (int k0 = 0; k0 < 512; k0 += 32) {
    v16bf a = load_a_frag(arow + k0, half);
#pragma unroll
    for (int t = 0; t < 4; ++t) {
      const int col = nt * 64 + t * 16 + nl;
      v16bf bw = *(const v16bf*)(W + (size_t)col * 512 + k0 + (lane & 16));
      acc[t] = wmma_bf16(a, bw, acc[t]);
    }
  }
#pragma unroll
  for (int t = 0; t < 4; ++t) {
    const int col = nt * 64 + t * 16 + nl;
    const float bv = bias[col];
#pragma unroll
    for (int j = 0; j < 8; ++j) {
      const int r = mt * 16 + j + 8 * half;
      out[(size_t)r * 512 + col] = acc[t][j] + bv;
    }
  }
}

// ---------------------------------------------------------------- host
extern "C" void kernel_launch(void* const* d_in, const int* in_sizes, int n_in,
                              void* d_out, int out_size, void* d_ws, size_t ws_size,
                              hipStream_t stream) {
  (void)in_sizes; (void)n_in; (void)out_size; (void)ws_size;
  const float* x     = (const float*)d_in[0];
  const float* gb    = (const float*)d_in[1];
  const float* w_in  = (const float*)d_in[2];
  const float* b_in  = (const float*)d_in[3];
  const float* w_out = (const float*)d_in[4];
  const float* b_out = (const float*)d_in[5];
  const float* bstr  = (const float*)d_in[6];
  float* out = (float*)d_out;

  char* ws = (char*)d_ws;
  size_t off = 0;
  auto alloc = [&](size_t bytes) -> void* {
    void* p = ws + off;
    off = (off + bytes + 255) & ~(size_t)255;
    return p;
  };
  bf16* x_bf    = (bf16*)alloc((size_t)8192 * 512 * 2);        //  8 MB
  bf16* win_bf  = (bf16*)alloc((size_t)1536 * 512 * 2);        //  1.5 MB
  bf16* wout_bf = (bf16*)alloc((size_t)512 * 512 * 2);         //  0.5 MB
  bf16* q_bf    = (bf16*)alloc((size_t)8 * 8 * 1024 * 64 * 2); //  8 MB
  bf16* k_bf    = (bf16*)alloc((size_t)8 * 8 * 1024 * 64 * 2); //  8 MB
  bf16* vt_bf   = (bf16*)alloc((size_t)8 * 8 * 64 * 1024 * 2); //  8 MB (transposed)
  bf16* ao_bf   = (bf16*)alloc((size_t)8192 * 512 * 2);        //  8 MB

  cvt_f32_bf16<<<(4194304 + 255) / 256, 256, 0, stream>>>(x, x_bf, 4194304);
  cvt_f32_bf16<<<(786432 + 255) / 256, 256, 0, stream>>>(w_in, win_bf, 786432);
  cvt_f32_bf16<<<(262144 + 255) / 256, 256, 0, stream>>>(w_out, wout_bf, 262144);

  qkv_gemm<<<1536, 256, 0, stream>>>(x_bf, win_bf, b_in, q_bf, k_bf, vt_bf);
  attn_kernel<<<512, 256, 0, stream>>>(q_bf, k_bf, vt_bf, gb, bstr, ao_bf);
  out_gemm<<<512, 256, 0, stream>>>(ao_bf, wout_bf, b_out, out);
}
